// Custom_Linear_PerRow_37941741093584
// MI455X (gfx1250) — 8-bit matrix operand order repaired and checked by reading, compiled, not run on hardware
//
#include <hip/hip_runtime.h>
#include <stdint.h>

typedef int v8i __attribute__((ext_vector_type(8)));

static constexpr int S_DIM = 8192;
static constexpr int K_DIM = 4096;
static constexpr int N_DIM = 4096;

static constexpr int      ROW_PITCH  = 80;
static constexpr uint32_t TILE_BYTES = 128 * ROW_PITCH;
static constexpr uint32_t BUF_BYTES  = 2 * TILE_BYTES;
static constexpr uint32_t NBUF       = 3;
static constexpr uint32_t LDS_BYTES  = NBUF * BUF_BYTES;

__device__ __forceinline__ void async_copy_b128(uint32_t lds_addr, const void* gaddr)
{
    asm volatile("global_load_async_to_lds_b128 %0, %1, off"
                 :: "v"(lds_addr), "v"(gaddr)
                 : "memory");
}

template <int N>
__device__ __forceinline__ void wait_asynccnt()
{
#if __has_builtin(__builtin_amdgcn_s_wait_asynccnt)
    __builtin_amdgcn_s_wait_asynccnt(N);
#else
    asm volatile("s_wait_asynccnt %0" :: "i"(N) : "memory");
#endif
}

__global__ __launch_bounds__(256)
void pack_i8_kernel(const int* __restrict__ src, int8_t* __restrict__ dst, int n4)
{
    int i = blockIdx.x * blockDim.x + threadIdx.x;
    if (i < n4) {
        const int4 v = reinterpret_cast<const int4*>(src)[i];
        char4 c;
        c.x = (signed char)v.x;
        c.y = (signed char)v.y;
        c.z = (signed char)v.z;
        c.w = (signed char)v.w;
        reinterpret_cast<char4*>(dst)[i] = c;
    }
}

__global__ __launch_bounds__(256)
void copy_scale_y_kernel(const float* __restrict__ sy, float* __restrict__ dst, int n)
{
    int i = blockIdx.x * blockDim.x + threadIdx.x;
    if (i < n) dst[i] = sy[i];
}

__global__ __launch_bounds__(256, 1)
void gemm_i8_wmma_kernel(const int8_t* __restrict__ xq,
                         const int8_t* __restrict__ wq,
                         const float* __restrict__ scale_x,
                         const float* __restrict__ scale_w,
                         const float* __restrict__ scale_y,
                         float* __restrict__ out)
{
    extern __shared__ int8_t smem[];

    const int t     = threadIdx.x;
    const int lane  = t & 31;
    const int wave  = t >> 5;
    const int waveM = wave >> 2;
    const int waveN = wave & 3;
    const int half  = lane >> 4;
    const int l16   = lane & 15;

    const int blockM = blockIdx.y * 128;
    const int blockN = blockIdx.x * 128;
    const int m0 = blockM + waveM * 64;
    const int n0 = blockN + waveN * 32;

    const int rChunk = t >> 2;
    const int cChunk = (t & 3) * 16;
    const int8_t* gA = xq + (size_t)(blockM + rChunk) * K_DIM + cChunk;
    const int8_t* gB = wq + (size_t)(blockN + rChunk) * K_DIM + cChunk;
    const size_t  g64 = (size_t)64 * K_DIM;

    const uint32_t ldsBase = (uint32_t)(size_t)smem;
    const uint32_t ldsA    = ldsBase + rChunk * ROW_PITCH + cChunk;
    const uint32_t ldsB    = ldsA + TILE_BYTES;
    const uint32_t row64   = 64 * ROW_PITCH;

    v8i acc[4][2];
#pragma unroll
    for (int mt = 0; mt < 4; ++mt)
#pragma unroll
        for (int nt = 0; nt < 2; ++nt)
            acc[mt][nt] = (v8i){0, 0, 0, 0, 0, 0, 0, 0};

    async_copy_b128(ldsA,         gA);
    async_copy_b128(ldsA + row64, gA + g64);
    async_copy_b128(ldsB,         gB);
    async_copy_b128(ldsB + row64, gB + g64);

    const int KSTEPS = K_DIM / 64;
    uint32_t curOff = 0;
#pragma unroll 1
    for (int it = 0; it < KSTEPS; ++it) {
        const uint32_t nxtOff =
            (curOff == (NBUF - 1) * BUF_BYTES) ? 0u : (curOff + BUF_BYTES);

        if (it + 1 < KSTEPS) {
            const int koff = (it + 1) * 64;
            async_copy_b128(ldsA + nxtOff,         gA + koff);
            async_copy_b128(ldsA + nxtOff + row64, gA + koff + g64);
            async_copy_b128(ldsB + nxtOff,         gB + koff);
            async_copy_b128(ldsB + nxtOff + row64, gB + koff + g64);
            wait_asynccnt<4>();
        } else {
            wait_asynccnt<0>();
        }
        __syncthreads();

        const int8_t* sAb = smem + curOff;
        const int8_t* sBb = smem + curOff + TILE_BYTES;

        v8i bfrag[2];
#pragma unroll
        for (int nt = 0; nt < 2; ++nt) {
            const int8_t* bp = sBb + (waveN * 32 + nt * 16 + l16) * ROW_PITCH + half * 16;
            const int4 lo = *reinterpret_cast<const int4*>(bp);
            const int4 hi = *reinterpret_cast<const int4*>(bp + 32);
            bfrag[nt] = (v8i){lo.x, lo.y, lo.z, lo.w, hi.x, hi.y, hi.z, hi.w};
        }
#pragma unroll
        for (int mt = 0; mt < 4; ++mt) {
            const int8_t* ap = sAb + (waveM * 64 + mt * 16 + l16) * ROW_PITCH + half * 16;
            const int4 q0 = *reinterpret_cast<const int4*>(ap +  0);
            const int4 q1 = *reinterpret_cast<const int4*>(ap + 32);
            const v8i afrag = (v8i){q0.x, q0.y, q0.z, q0.w, q1.x, q1.y, q1.z, q1.w};

            acc[mt][0] = __builtin_amdgcn_wmma_i32_16x16x64_iu8(
                true, afrag, true, bfrag[0], acc[mt][0], false, false);
            acc[mt][1] = __builtin_amdgcn_wmma_i32_16x16x64_iu8(
                true, afrag, true, bfrag[1], acc[mt][1], false, false);
        }
        curOff = nxtOff;
    }

#pragma unroll
    for (int mt = 0; mt < 4; ++mt) {
        float sxy[8];
#pragma unroll
        for (int r = 0; r < 8; ++r) {
            const int s = m0 + mt * 16 + half * 8 + r;
            sxy[r] = scale_x[s] / scale_y[s];
        }
#pragma unroll
        for (int nt = 0; nt < 2; ++nt) {
            const int n = n0 + nt * 16 + l16;
            const float sw = scale_w[n];
#pragma unroll
            for (int r = 0; r < 8; ++r) {
                const int s = m0 + mt * 16 + half * 8 + r;
                float f = (float)acc[mt][nt][r] * (sxy[r] * sw);
                f = __builtin_rintf(f);
                f = fminf(127.0f, fmaxf(-128.0f, f));
                out[(size_t)s * N_DIM + n] = f;
            }
        }
    }
}

extern "C" void kernel_launch(void* const* d_in, const int* in_sizes, int n_in,
                              void* d_out, int out_size, void* d_ws, size_t ws_size,
                              hipStream_t stream)
{
    const int*   x_i32 = (const int*)d_in[0];
    const int*   w_i32 = (const int*)d_in[1];
    const float* sx    = (const float*)d_in[2];
    const float* sw    = (const float*)d_in[3];
    const float* sy    = (const float*)d_in[4];
    float*       out   = (float*)d_out;

    int8_t* xq = (int8_t*)d_ws;
    int8_t* wq = xq + (size_t)S_DIM * K_DIM;

    {
        const int n4 = (S_DIM * K_DIM) / 4;
        pack_i8_kernel<<<(n4 + 255) / 256, 256, 0, stream>>>(x_i32, xq, n4);
    }
    {
        const int n4 = (N_DIM * K_DIM) / 4;
        pack_i8_kernel<<<(n4 + 255) / 256, 256, 0, stream>>>(w_i32, wq, n4);
    }

    dim3 grid(N_DIM / 128, S_DIM / 128);
    gemm_i8_wmma_kernel<<<grid, dim3(256), LDS_BYTES, stream>>>(xq, wq, sx, sw, sy, out);

    copy_scale_y_kernel<<<(S_DIM + 255) / 256, 256, 0, stream>>>(
        sy, out + (size_t)S_DIM * N_DIM, S_DIM);
}
